// TransformerLayer_38079180046888
// MI455X (gfx1250) — matrix operand order repaired and checked by reading, not compiled, not run on hardware
//
#include <hip/hip_runtime.h>
#include <hip/hip_bf16.h>
#include <math.h>

// ---------------------------------------------------------------------------
// Types
// ---------------------------------------------------------------------------
typedef __bf16 bf16_t;
typedef __attribute__((ext_vector_type(16))) __bf16 v16bf;
typedef __attribute__((ext_vector_type(8)))  float  v8f;
typedef __attribute__((ext_vector_type(4)))  unsigned int u32x4;
typedef __attribute__((ext_vector_type(8)))  int i32x8;
typedef __attribute__((ext_vector_type(4)))  int i32x4;

#define WMMA_BF16(a, b, c) \
  __builtin_amdgcn_wmma_f32_16x16x32_bf16(false, (a), false, (b), (short)0, (c), false, false)

// Force the TDM path on the device pass: tensor_load_to_lds is probe-confirmed
// on both known gfx1250 toolchains. Host pass keeps the fallback branch so the
// kernel body still parses host-side.
#if defined(__gfx1250__)
#define USE_TDM 1
#else
#define USE_TDM 0
#endif

__device__ __forceinline__ v8f v8f_zero() {
  v8f z = {0.f, 0.f, 0.f, 0.f, 0.f, 0.f, 0.f, 0.f};
  return z;
}

// Load 16 bf16 as two 16-byte chunks (chunk0 -> elems 0..7, chunk1 -> elems 8..15)
__device__ __forceinline__ v16bf ldpair(const bf16_t* c0, const bf16_t* c1) {
  v16bf r;
  reinterpret_cast<uint4*>(&r)[0] = *reinterpret_cast<const uint4*>(c0);
  reinterpret_cast<uint4*>(&r)[1] = *reinterpret_cast<const uint4*>(c1);
  return r;
}

// A fragment (16 rows x 32 K) from row-major global memory, leading dim ld.
// lane<16: M=lane, K={k0..k0+7}u{k0+16..k0+23}; lane>=16: M=lane-16, K={k0+8..}u{k0+24..}
__device__ __forceinline__ v16bf load_a_frag(const bf16_t* __restrict__ A, int ld,
                                             int row0, int k0, int lane) {
  int m  = row0 + (lane & 15);
  int ks = k0 + ((lane >> 4) << 3);
  const bf16_t* p = A + (size_t)m * ld + ks;
  return ldpair(p, p + 16);
}

// B fragment for y = x @ W^T (B[k,n] = W[n,k]); W row-major [N,K].
__device__ __forceinline__ v16bf load_b_fragT(const bf16_t* __restrict__ W, int ld,
                                              int n0, int k0, int lane) {
  int n  = n0 + (lane & 15);
  int ks = k0 + ((lane >> 4) << 3);
  const bf16_t* p = W + (size_t)n * ld + ks;
  return ldpair(p, p + 16);
}

// Same fragments out of LDS tiles with padded row stride 40 bf16 (32 data + 8 pad).
__device__ __forceinline__ v16bf lds_a_frag(const bf16_t* S, int row0, int lane) {
  int m  = row0 + (lane & 15);
  int ks = (lane >> 4) << 3;
  const bf16_t* p = S + m * 40 + ks;
  return ldpair(p, p + 16);
}
__device__ __forceinline__ v16bf lds_b_frag(const bf16_t* S, int row0, int lane) {
  int n  = row0 + (lane & 15);
  int ks = (lane >> 4) << 3;
  const bf16_t* p = S + n * 40 + ks;
  return ldpair(p, p + 16);
}

#if USE_TDM
// Issue one TDM 2D tile load: rows x 32 bf16 tile from row-major src (leading dim
// ld elements) into LDS at lds_off, padded 16B after every 64B row -> stride 80B.
__device__ __forceinline__ void tdm_load_tile(const bf16_t* gsrc, unsigned lds_off,
                                              int rows, int ld) {
  unsigned long long ga = (unsigned long long)(uintptr_t)gsrc;
  u32x4 g0;
  g0[0] = 1u;                                              // count=1, user mode
  g0[1] = lds_off;                                         // lds_addr (bytes)
  g0[2] = (unsigned)(ga & 0xffffffffull);                  // global_addr[31:0]
  g0[3] = (unsigned)((ga >> 32) & 0x01ffffffull) | (2u << 30);  // addr[56:32] | type=2
  i32x8 g1;
  // data_size=1(2B)<<16, pad_enable<<20, pad_interval=3 (16 DW)<<22, pad_amount=3 (4 DW)<<25
  g1[0] = (int)((1u << 16) | (1u << 20) | (3u << 22) | (3u << 25));
  g1[1] = (int)(32u << 16);                                // tensor_dim0[15:0] = 32
  g1[2] = (int)(((unsigned)rows & 0xffffu) << 16);         // tensor_dim1[15:0] = rows
  g1[3] = (int)(32u << 16);                                // tile_dim0 = 32
  g1[4] = (int)((unsigned)rows & 0xffffu);                 // tile_dim1 = rows, tile_dim2=0
  g1[5] = (int)(unsigned)ld;                               // tensor_dim0_stride[31:0]
  g1[6] = 0;
  g1[7] = 0;
  i32x4 z4 = {0, 0, 0, 0};
#if __clang_major__ >= 23
  i32x8 z8 = {0, 0, 0, 0, 0, 0, 0, 0};
  __builtin_amdgcn_tensor_load_to_lds(g0, g1, z4, z4, z8, 0);
#else
  __builtin_amdgcn_tensor_load_to_lds(g0, g1, z4, z4, 0);
#endif
}
#endif

// ---------------------------------------------------------------------------
// fp32 -> bf16 elementwise cast (weights)
// ---------------------------------------------------------------------------
__global__ void cast_f32_bf16_k(const float* __restrict__ src, bf16_t* __restrict__ dst, int n) {
  int i = blockIdx.x * blockDim.x + threadIdx.x;
  if (i < n) dst[i] = (bf16_t)src[i];
}

// ---------------------------------------------------------------------------
// LayerNorm (D=1024) + cast to bf16. One block (256 thr) per row.
// ---------------------------------------------------------------------------
__global__ __launch_bounds__(256) void ln_cast_k(const float* __restrict__ x,
                                                 const float* __restrict__ w,
                                                 const float* __restrict__ b,
                                                 bf16_t* __restrict__ out) {
  __shared__ float red[256];
  const int row = blockIdx.x;
  const float* xr = x + (size_t)row * 1024;
  float v[4];
  float s = 0.f;
#pragma unroll
  for (int i = 0; i < 4; ++i) { v[i] = xr[threadIdx.x + 256 * i]; s += v[i]; }
  red[threadIdx.x] = s; __syncthreads();
  for (int off = 128; off > 0; off >>= 1) {
    if (threadIdx.x < off) red[threadIdx.x] += red[threadIdx.x + off];
    __syncthreads();
  }
  const float mean = red[0] * (1.0f / 1024.0f);
  __syncthreads();
  float s2 = 0.f;
#pragma unroll
  for (int i = 0; i < 4; ++i) { float d = v[i] - mean; s2 += d * d; }
  red[threadIdx.x] = s2; __syncthreads();
  for (int off = 128; off > 0; off >>= 1) {
    if (threadIdx.x < off) red[threadIdx.x] += red[threadIdx.x + off];
    __syncthreads();
  }
  const float inv = rsqrtf(red[0] * (1.0f / 1024.0f) + 1e-5f);
#pragma unroll
  for (int i = 0; i < 4; ++i) {
    int c = threadIdx.x + 256 * i;
    out[(size_t)row * 1024 + c] = (bf16_t)((v[i] - mean) * inv * w[c] + b[c]);
  }
}

// ---------------------------------------------------------------------------
// Block-tiled bf16 WMMA GEMM: C = A[M,K] @ W[N,K]^T.
// Block = 8 waves (2Mx4N), block tile 64x256, K-step 32.
// A/B tiles double-buffered in LDS (padded stride 40), loaded by TDM
// (tensor_load_to_lds, TENSORcnt). EPI 0: QKV scatter; EPI 1: attn-out +
// residual; EPI 2: bias+GELU; EPI 3: bias+residual.
// ---------------------------------------------------------------------------
template <int EPI>
__global__ __launch_bounds__(256) void gemm_bf16_wmma_k(
    const bf16_t* __restrict__ A, const bf16_t* __restrict__ W,
    const float* __restrict__ bias, const float* __restrict__ res,
    float* __restrict__ outf, bf16_t* __restrict__ ob0,
    bf16_t* __restrict__ ob1, bf16_t* __restrict__ ob2,
    int M, int N, int K, float scale) {
  __shared__ __align__(16) bf16_t As[2][64 * 40];
  __shared__ __align__(16) bf16_t Bs[2][256 * 40];

  const int tid  = threadIdx.x;
  const int lane = tid & 31;
  const int wv   = tid >> 5;
  const int nblk = N >> 8;                 // N / 256
  const int bm   = blockIdx.x / nblk;
  const int bn   = blockIdx.x - bm * nblk;
  const int gm0  = bm << 6;                // block row base (64 rows)
  const int gn0  = bn << 8;                // block col base (256 cols)
  const int m_off = (wv >> 2) << 5;        // 0 / 32
  const int n_off = (wv & 3) << 6;         // 0 / 64 / 128 / 192

  v8f acc[2][4];
#pragma unroll
  for (int f = 0; f < 2; ++f)
#pragma unroll
    for (int j = 0; j < 4; ++j) acc[f][j] = v8f_zero();

  auto stage = [&](int buf, int k0) {
#if USE_TDM
    if (wv == 0) {
      tdm_load_tile(W + (size_t)gn0 * K + k0,
                    (unsigned)(uintptr_t)(const void*)&Bs[buf][0], 256, K);
      tdm_load_tile(A + (size_t)gm0 * K + k0,
                    (unsigned)(uintptr_t)(const void*)&As[buf][0], 64, K);
    }
#else
#pragma unroll
    for (int i = 0; i < 4; ++i) {
      int ch = tid + (i << 8);
      int row = ch >> 2, c = (ch & 3) << 3;
      *reinterpret_cast<uint4*>(&Bs[buf][row * 40 + c]) =
          *reinterpret_cast<const uint4*>(&W[(size_t)(gn0 + row) * K + k0 + c]);
    }
    {
      int row = tid >> 2, c = (tid & 3) << 3;
      *reinterpret_cast<uint4*>(&As[buf][row * 40 + c]) =
          *reinterpret_cast<const uint4*>(&A[(size_t)(gm0 + row) * K + k0 + c]);
    }
#endif
  };

  stage(0, 0);
  for (int k0 = 0; k0 < K; k0 += 32) {
    const int buf = (k0 >> 5) & 1;
#if USE_TDM
    if (wv == 0) __builtin_amdgcn_s_wait_tensorcnt(0);
#endif
    __syncthreads();
    if (k0 + 32 < K) stage(buf ^ 1, k0 + 32);
    v16bf a0 = lds_a_frag(&As[buf][0], m_off, lane);
    v16bf a1 = lds_a_frag(&As[buf][0], m_off + 16, lane);
#pragma unroll
    for (int j = 0; j < 4; ++j) {
      v16bf bfr = lds_b_frag(&Bs[buf][0], n_off + j * 16, lane);
      acc[0][j] = WMMA_BF16(a0, bfr, acc[0][j]);
      acc[1][j] = WMMA_BF16(a1, bfr, acc[1][j]);
    }
  }

  const int lhalf = (lane >> 4) << 3;
  const int lcol  = lane & 15;
#pragma unroll
  for (int f = 0; f < 2; ++f) {
#pragma unroll
    for (int j = 0; j < 4; ++j) {
#pragma unroll
      for (int r = 0; r < 8; ++r) {
        const int m = gm0 + m_off + f * 16 + r + lhalf;
        const int n = gn0 + n_off + j * 16 + lcol;
        float v = acc[f][j][r];
        if constexpr (EPI == 0) {
          v *= scale;
          const int d  = n / 48;
          const int rm = n - d * 48;
          const int z  = rm >> 4;
          const int hh = rm & 15;
          const int b_ = m >> 11;          // S = 2048
          const int s_ = m & 2047;
          if (z == 2) {                    // V stored transposed: [bh][d][s]
            ob2[(((size_t)(b_ * 16 + hh) * 64) + d) * 2048 + s_] = (bf16_t)v;
          } else {
            const size_t idx = (((size_t)(b_ * 16 + hh) * 2048) + s_) * 64 + d;
            if (z == 0) ob0[idx] = (bf16_t)v; else ob1[idx] = (bf16_t)v;
          }
        } else if constexpr (EPI == 1) {
          v *= scale;
          const size_t idx = (size_t)m * N + n;
          outf[idx] = 0.44721360f * v + 0.89442719f * res[idx];
        } else if constexpr (EPI == 2) {
          v = (v + bias[n]) * scale;
          const float g = 0.5f * v * (1.0f + erff(v * 0.70710678f));
          ob0[(size_t)m * N + n] = (bf16_t)(g * 1.58730159f);
        } else {
          v = (v + bias[n]) * scale;
          const size_t idx = (size_t)m * N + n;
          outf[idx] = 0.44721360f * v + 0.89442719f * res[idx];
        }
      }
    }
  }
}

// ---------------------------------------------------------------------------
// Flash attention. One wave = one (b,h, 16-row q tile); 4 waves/block.
// 64-key chunks: 8 score WMMAs + online softmax + LDS relayout + 8 AV WMMAs.
// Q/K: [bh][s][64] bf16; Vt: [bh][d][s] bf16; AV out: [b][s][h*64+d] bf16.
// ---------------------------------------------------------------------------
__global__ __launch_bounds__(128) void attn_flash_k(
    const bf16_t* __restrict__ Q, const bf16_t* __restrict__ Kb,
    const bf16_t* __restrict__ Vt, bf16_t* __restrict__ AV) {
  __shared__ __align__(16) bf16_t Pbuf[4][16][72];   // 64 data + 8 pad per row
  const int lane = threadIdx.x & 31;
  const int w    = threadIdx.x >> 5;
  const int wg   = blockIdx.x * 4 + w;
  const int bh   = wg >> 7;
  const int qt   = wg & 127;
  const int q0   = qt << 4;

  const bf16_t* Qp = Q  + (size_t)bh * 2048 * 64;
  const bf16_t* Kp = Kb + (size_t)bh * 2048 * 64;
  const bf16_t* Vp = Vt + (size_t)bh * 64 * 2048;

  const v16bf qA0 = load_a_frag(Qp, 64, q0, 0, lane);
  const v16bf qA1 = load_a_frag(Qp, 64, q0, 32, lane);

  v8f o[4];
#pragma unroll
  for (int j = 0; j < 4; ++j) o[j] = v8f_zero();
  float mrun[8], lrun[8];
#pragma unroll
  for (int r = 0; r < 8; ++r) { mrun[r] = -3.0e38f; lrun[r] = 0.0f; }

  const int lhalf = (lane >> 4) << 3;
  const int lcol  = lane & 15;
  const float MM_SCALE = 0.03937253f;  // (S*S*DH)^(-1/6)

  for (int t0 = 0; t0 < 2048; t0 += 64) {
    float p[4][8];
#pragma unroll
    for (int ct = 0; ct < 4; ++ct) {
      v8f acc = v8f_zero();
      v16bf kb = load_b_fragT(Kp, 64, t0 + ct * 16, 0, lane);
      acc = WMMA_BF16(qA0, kb, acc);
      kb = load_b_fragT(Kp, 64, t0 + ct * 16, 32, lane);
      acc = WMMA_BF16(qA1, kb, acc);
#pragma unroll
      for (int r = 0; r < 8; ++r) p[ct][r] = acc[r] * MM_SCALE;
    }
#pragma unroll
    for (int r = 0; r < 8; ++r) {
      float t = fmaxf(fmaxf(p[0][r], p[1][r]), fmaxf(p[2][r], p[3][r]));
      t = fmaxf(t, __shfl_xor(t, 1));
      t = fmaxf(t, __shfl_xor(t, 2));
      t = fmaxf(t, __shfl_xor(t, 4));
      t = fmaxf(t, __shfl_xor(t, 8));        // stays within each 16-lane half
      const float mnew  = fmaxf(mrun[r], t);
      const float alpha = __expf(mrun[r] - mnew);
      mrun[r] = mnew;
      lrun[r] *= alpha;
#pragma unroll
      for (int j = 0; j < 4; ++j) o[j][r] *= alpha;
      float su = 0.f;
#pragma unroll
      for (int ct = 0; ct < 4; ++ct) {
        const float e = __expf(p[ct][r] - mnew);
        p[ct][r] = e;
        su += e;
      }
      su += __shfl_xor(su, 1);
      su += __shfl_xor(su, 2);
      su += __shfl_xor(su, 4);
      su += __shfl_xor(su, 8);
      lrun[r] += su;
    }
#pragma unroll
    for (int r = 0; r < 8; ++r) {
#pragma unroll
      for (int ct = 0; ct < 4; ++ct)
        Pbuf[w][r + lhalf][ct * 16 + lcol] = (bf16_t)p[ct][r];
    }
    __syncthreads();
    const int prow = lane & 15;
    const int pks  = (lane >> 4) << 3;
    const v16bf pA0 = ldpair(&Pbuf[w][prow][pks], &Pbuf[w][prow][pks + 16]);
    const v16bf pA1 = ldpair(&Pbuf[w][prow][32 + pks], &Pbuf[w][prow][32 + pks + 16]);
    __syncthreads();
#pragma unroll
    for (int j = 0; j < 4; ++j) {
      const int d = j * 16 + lcol;
      const bf16_t* vp = Vp + (size_t)d * 2048 + t0 + ((lane >> 4) << 3);
      o[j] = WMMA_BF16(pA0, ldpair(vp, vp + 16), o[j]);
      const bf16_t* vp2 = vp + 32;
      o[j] = WMMA_BF16(pA1, ldpair(vp2, vp2 + 16), o[j]);
    }
  }

  const int b_ = bh >> 4, hh = bh & 15;
#pragma unroll
  for (int r = 0; r < 8; ++r) {
    const float inv = 54.8461f / lrun[r];  // SM_SCALE * MM_SCALE
    const int s_ = q0 + r + lhalf;
    const size_t rowoff = ((size_t)(b_ * 2048) + s_) * 1024 + hh * 64;
#pragma unroll
    for (int j = 0; j < 4; ++j)
      AV[rowoff + j * 16 + lcol] = (bf16_t)(o[j][r] * inv);
  }
}

// ---------------------------------------------------------------------------
// Host orchestration
// ---------------------------------------------------------------------------
extern "C" void kernel_launch(void* const* d_in, const int* in_sizes, int n_in,
                              void* d_out, int out_size, void* d_ws, size_t ws_size,
                              hipStream_t stream) {
  const float* x     = (const float*)d_in[0];
  const float* w_qkv = (const float*)d_in[1];
  const float* w_o   = (const float*)d_in[2];
  const float* w1    = (const float*)d_in[3];
  const float* b1    = (const float*)d_in[4];
  const float* w2    = (const float*)d_in[5];
  const float* b2    = (const float*)d_in[6];
  const float* ln1w  = (const float*)d_in[7];
  const float* ln1b  = (const float*)d_in[8];
  const float* ln2w  = (const float*)d_in[9];
  const float* ln2b  = (const float*)d_in[10];
  float* out = (float*)d_out;

  char* ws = (char*)d_ws;
  auto carve = [&](size_t bytes) -> char* {
    char* p = ws;
    ws += (bytes + 255) & ~(size_t)255;
    return p;
  };
  bf16_t* wqkv_b = (bf16_t*)carve((size_t)3072 * 1024 * 2);
  bf16_t* wo_b   = (bf16_t*)carve((size_t)1024 * 1024 * 2);
  bf16_t* w1_b   = (bf16_t*)carve((size_t)4096 * 1024 * 2);
  bf16_t* w2_b   = (bf16_t*)carve((size_t)4096 * 1024 * 2);
  bf16_t* h_b    = (bf16_t*)carve((size_t)4096 * 1024 * 2);   // LN out (reused for LN2)
  bf16_t* Qb     = (bf16_t*)carve((size_t)32 * 2048 * 64 * 2);
  bf16_t* Kbuf   = (bf16_t*)carve((size_t)32 * 2048 * 64 * 2);
  bf16_t* Vtb    = (bf16_t*)carve((size_t)32 * 64 * 2048 * 2);
  bf16_t* av_b   = (bf16_t*)carve((size_t)4096 * 1024 * 2);
  bf16_t* mid_b  = (bf16_t*)carve((size_t)4096 * 4096 * 2);
  float*  x1     = (float*)carve((size_t)4096 * 1024 * 4);

  const float QKV_SCALE = 0.02374487f;   // (1024*3072)^-0.25
  const float WO_SCALE  = 0.03125f;      // (1024*1024)^-0.25
  const float W1_SCALE  = 0.02209709f;   // (1024*4096)^-0.25
  const float W2_SCALE  = 0.02209709f;

  // 1) Weights -> bf16
  cast_f32_bf16_k<<<(3145728 + 255) / 256, 256, 0, stream>>>(w_qkv, wqkv_b, 3145728);
  cast_f32_bf16_k<<<(1048576 + 255) / 256, 256, 0, stream>>>(w_o, wo_b, 1048576);
  cast_f32_bf16_k<<<(4194304 + 255) / 256, 256, 0, stream>>>(w1, w1_b, 4194304);
  cast_f32_bf16_k<<<(4194304 + 255) / 256, 256, 0, stream>>>(w2, w2_b, 4194304);

  // 2) LN1 -> bf16 h
  ln_cast_k<<<4096, 256, 0, stream>>>(x, ln1w, ln1b, h_b);

  // 3) QKV projection: blocks = (4096/64)*(3072/256) = 64*12 = 768
  gemm_bf16_wmma_k<0><<<768, 256, 0, stream>>>(h_b, wqkv_b, nullptr, nullptr,
                                               nullptr, Qb, Kbuf, Vtb,
                                               4096, 3072, 1024, QKV_SCALE);

  // 4) Flash attention: 4096 waves -> 1024 blocks of 4 waves
  attn_flash_k<<<1024, 128, 0, stream>>>(Qb, Kbuf, Vtb, av_b);

  // 5) Output projection + residual: 64*4 = 256 blocks
  gemm_bf16_wmma_k<1><<<256, 256, 0, stream>>>(av_b, wo_b, nullptr, x,
                                               x1, nullptr, nullptr, nullptr,
                                               4096, 1024, 1024, WO_SCALE);

  // 6) LN2 -> bf16 h (reuse h_b)
  ln_cast_k<<<4096, 256, 0, stream>>>(x1, ln2w, ln2b, h_b);

  // 7) MLP up + GELU: 64*16 = 1024 blocks
  gemm_bf16_wmma_k<2><<<1024, 256, 0, stream>>>(h_b, w1_b, b1, nullptr,
                                                nullptr, mid_b, nullptr, nullptr,
                                                4096, 4096, 1024, W1_SCALE);

  // 8) MLP down + residual -> d_out: 64*4 = 256 blocks
  gemm_bf16_wmma_k<3><<<256, 256, 0, stream>>>(mid_b, w2_b, b2, x1,
                                               out, nullptr, nullptr, nullptr,
                                               4096, 1024, 4096, W2_SCALE);
}
